// GrapgRerank_37125697306746
// MI455X (gfx1250) — matrix operand order repaired and checked by reading, not compiled, not run on hardware
//
#include <hip/hip_runtime.h>

// ---------------- problem constants ----------------
#define BB   16
#define NCC  12
#define LL   500
#define DD   256
constexpr int   NROWS  = BB * NCC * LL;   // 96000
constexpr float TINV   = 2.0f;            // 1/T, T = 0.5
constexpr float LN_EPS = 1e-6f;

typedef __attribute__((ext_vector_type(16))) _Float16 v16h;
typedef __attribute__((ext_vector_type(8)))  _Float16 v8h;
typedef __attribute__((ext_vector_type(4)))  _Float16 h4;
typedef __attribute__((ext_vector_type(8)))  float    v8f;
typedef __attribute__((ext_vector_type(4)))  float    fx4;

__device__ inline v8f v8zero() {
  v8f z;
#pragma unroll
  for (int i = 0; i < 8; ++i) z[i] = 0.f;
  return z;
}
__device__ inline v8h h8zero() {
  v8h z;
#pragma unroll
  for (int i = 0; i < 8; ++i) z[i] = (_Float16)0.f;
  return z;
}

__device__ inline float geluf(float x) {
  return 0.5f * x * (1.f + erff(x * 0.70710678118654752f));
}
__device__ inline float clamp01(float a) { return fminf(fmaxf(a, 0.1f), 0.9f); }

// ---- CDNA5 async global->LDS copy (ASYNCcnt path), 16 bytes per op -------
__device__ inline void async_b128(const _Float16* ldsPtr, const _Float16* gptr) {
  unsigned loff = (unsigned)(uintptr_t)ldsPtr;                // LDS byte offset
  unsigned long long ga = (unsigned long long)(uintptr_t)gptr;
  asm volatile("global_load_async_to_lds_b128 %0, %1, off"
               :: "v"(loff), "v"(ga) : "memory");
}
__device__ inline void async_wait0() {
  asm volatile("s_wait_asynccnt 0x0" ::: "memory");
}

// ---------------- WMMA fragment loaders (ISA 7.12.2, wave32) --------------
__device__ inline v16h frag16(v8h lo, v8h hi) {
  return __builtin_shufflevector(lo, hi, 0, 1, 2, 3, 4, 5, 6, 7,
                                         8, 9, 10, 11, 12, 13, 14, 15);
}
// A 16x32 f16, row-major tile: lane l: m=l&15, runs [kb..kb+7],[kb+16..kb+23], kb=(l>=16)?8:0
__device__ inline v16h load_a_frag_v(const _Float16* lds, int row0, int k0, int stride) {
  int l = threadIdx.x & 31;
  const _Float16* base = lds + (row0 + (l & 15)) * stride + k0 + ((l >= 16) ? 8 : 0);
  return frag16(*(const v8h*)base, *(const v8h*)(base + 16));
}
// B 32x16 f16 from K-contiguous layout lds[sel][k]: lane l: sel=sel0+(l&15),
// runs [kb..kb+7],[kb+16..kb+23], kb=(l>=16)?8:0
__device__ inline v16h load_b_fragK(const _Float16* lds, int sel0, int koff, int stride) {
  int l = threadIdx.x & 31;
  const _Float16* base = lds + (sel0 + (l & 15)) * stride + koff + ((l >= 16) ? 8 : 0);
  return frag16(*(const v8h*)base, *(const v8h*)(base + 16));
}

#define WMMA_F32_F16(a, b, c) \
  __builtin_amdgcn_wmma_f32_16x16x32_f16(false, (a), false, (b), (short)0, (c), false, false)

// ==========================================================================
// Kernel 0: W (f32, KxN) -> Wt (f16, transposed NxK) for async B-tile copies
// ==========================================================================
__global__ __launch_bounds__(256)
void wcvt_kernel(const float* __restrict__ W, _Float16* __restrict__ Wt) {
  int i = blockIdx.x * 256 + threadIdx.x;      // 65536 total
  int n = i >> 8, k = i & 255;
  Wt[(size_t)n * DD + k] = (_Float16)W[(size_t)k * DD + n];
}

// ==========================================================================
// Kernel 1: Y = X @ W + bias  (M=96000, K=N=256), f16 WMMA, f32 accumulate
// B tiles arrive via global_load_async_to_lds_b128 from pre-transposed Wt.
// ==========================================================================
__global__ __launch_bounds__(256)
void gemm_xw_kernel(const float* __restrict__ X, const _Float16* __restrict__ Wt,
                    const float* __restrict__ bias, float* __restrict__ Y) {
  __shared__ __align__(16) _Float16 sA[32 * 32];      // X tile, row-major
  __shared__ __align__(16) _Float16 sBT[DD * 32];     // W tile, [n][k] K-contiguous

  const int m0   = blockIdx.x * 32;
  const int tid  = threadIdx.x;
  const int wave = tid >> 5;
  const int lane = tid & 31;
  const int msub = wave & 1;
  const int ngrp = wave >> 1;

  v8f acc[4];
#pragma unroll
  for (int t = 0; t < 4; ++t) acc[t] = v8zero();

  for (int kt = 0; kt < DD; kt += 32) {
    // A tile: f32 -> f16, vectorized (one fx4 load + h4 store per thread)
    {
      int r = tid >> 3, k4 = (tid & 7) << 2;
      fx4 f = *(const fx4*)&X[(size_t)(m0 + r) * DD + kt + k4];
      h4 h;
#pragma unroll
      for (int j = 0; j < 4; ++j) h[j] = (_Float16)f[j];
      *(h4*)&sA[r * 32 + k4] = h;
    }
    // B tile: async copy 256 rows x 64B from Wt[n][kt..kt+31]
    for (int i = tid; i < 1024; i += 256) {
      int n = i >> 2, c8 = (i & 3) << 3;
      async_b128(&sBT[n * 32 + c8], Wt + (size_t)n * DD + kt + c8);
    }
    if (kt + 32 < DD)
      __builtin_prefetch(&X[(size_t)(m0 + (tid >> 3)) * DD + kt + 32], 0, 3);
    async_wait0();
    __syncthreads();

    v16h a = load_a_frag_v(sA, msub * 16, 0, 32);
#pragma unroll
    for (int t = 0; t < 4; ++t) {
      v16h b = load_b_fragK(sBT, (ngrp * 4 + t) * 16, 0, 32);
      acc[t] = WMMA_F32_F16(a, b, acc[t]);
    }
    __syncthreads();
  }

  const int n  = lane & 15;
  const int mb = (lane >= 16) ? 8 : 0;
#pragma unroll
  for (int t = 0; t < 4; ++t) {
    int col = (ngrp * 4 + t) * 16 + n;
    float bv = bias[col];
#pragma unroll
    for (int v = 0; v < 8; ++v) {
      int row = m0 + msub * 16 + mb + v;
      Y[(size_t)row * DD + col] = acc[t][v] + bv;
    }
  }
}

// ==========================================================================
// Kernel 2: row squared norms of Y + f16 copy Yh (feeds async tile copies)
// ==========================================================================
__global__ __launch_bounds__(256)
void rownorm_kernel(const float* __restrict__ Y, float* __restrict__ y2,
                    _Float16* __restrict__ Yh) {
  int row  = blockIdx.x * 8 + (threadIdx.x >> 5);
  int lane = threadIdx.x & 31;
  if (row >= NROWS) return;
  float s = 0.f;
  for (int c = lane; c < DD; c += 32) {
    float v = Y[(size_t)row * DD + c];
    s += v * v;
    Yh[(size_t)row * DD + c] = (_Float16)v;
  }
#pragma unroll
  for (int off = 16; off; off >>= 1) s += __shfl_xor(s, off, 32);
  if (lane == 0) y2[row] = s;
}

// ==========================================================================
// Kernel 3: flash distance-softmax attention + LN + GELU + residual blend
// block = 256 threads (8 waves), one (b,c) x 16 query rows; 512-row key loop
// ==========================================================================
__global__ __launch_bounds__(256)
void attn_kernel(const _Float16* __restrict__ Yh, const float* __restrict__ y2,
                 const float* __restrict__ Xin, float* __restrict__ Xout,
                 const float* __restrict__ lng, const float* __restrict__ lnb,
                 const float* __restrict__ alphaP, const float* __restrict__ lamdaP,
                 int cross) {
  __shared__ __align__(16) _Float16 sQ[16 * DD];     // row-major (A operand)
  __shared__ __align__(16) _Float16 sKV[32 * DD];    // row-major (QK B^T source)
  __shared__ __align__(16) _Float16 sKVT[DD * 32];   // [col][keyrow] (PV B source)
  __shared__ __align__(16) _Float16 sP[16 * 32];     // probabilities (A operand)
  __shared__ float sq2[16], sk2[32], srs[16];
  __shared__ float sH[16 * DD];

  const int bid   = blockIdx.x;
  const int bc    = bid >> 5;
  const int qb    = bid & 31;
  const int b     = bc / NCC;
  const int qrow0 = qb * 16;
  const int qsrc  = cross ? (b * NCC) : bc;   // cross: queries from candidate 0
  const int tid   = threadIdx.x;
  const int wave  = tid >> 5;
  const int lane  = tid & 31;
  const float lam = lamdaP[0];
  const v8h z8 = h8zero();

  // ---- Q tile via async copy (f16, byte-exact); pad rows zero-filled
  for (int i = tid; i < 512; i += 256) {
    int r = i >> 5, c8 = (i & 31) << 3;
    int qr = qrow0 + r;
    if (qr < LL) async_b128(&sQ[r * DD + c8], Yh + ((size_t)qsrc * LL + qr) * DD + c8);
    else         *(v8h*)&sQ[r * DD + c8] = z8;
  }
  if (tid < 16) {
    int qr = qrow0 + tid;
    sq2[tid] = (qr < LL) ? y2[(size_t)qsrc * LL + qr] : 0.f;
    srs[tid] = 0.f;
  }
  async_wait0();
  __syncthreads();

  // hoist loop-invariant Q fragments (waves 0,1 only use them)
  v16h qa[8];
  if (wave < 2) {
#pragma unroll
    for (int kk = 0; kk < 8; ++kk) qa[kk] = load_a_frag_v(sQ, 0, kk * 32, DD);
  }

  v8f acc0 = v8zero(), acc1 = v8zero();

  for (int kt = 0; kt < 16; ++kt) {
    const int krow0 = kt * 32;
    // ---- KV tile via async copy
    for (int i = tid; i < 1024; i += 256) {
      int r = i >> 5, c8 = (i & 31) << 3;
      int kr = krow0 + r;
      if (kr < LL) async_b128(&sKV[r * DD + c8], Yh + ((size_t)bc * LL + kr) * DD + c8);
      else         *(v8h*)&sKV[r * DD + c8] = z8;
    }
    if (tid < 32) {
      int kr = krow0 + tid;
      sk2[tid] = (kr < LL) ? y2[(size_t)bc * LL + kr] : 0.f;
    }
    async_wait0();
    __syncthreads();

    // ---- transposed copy for the PV B operand (K-contiguous per column)
    for (int i = tid; i < 2048; i += 256) {
      int r = i >> 6, c4 = (i & 63) << 2;
      h4 v = *(const h4*)&sKV[r * DD + c4];
#pragma unroll
      for (int j = 0; j < 4; ++j) sKVT[(c4 + j) * 32 + r] = v[j];
    }
    __syncthreads();

    // ---- waves 0,1: S = Q @ K^T (one 16x16 tile each), logits -> P
    if (wave < 2) {
      v8f sacc = v8zero();
#pragma unroll
      for (int kk = 0; kk < 8; ++kk) {
        v16h bT = load_b_fragK(sKV, wave * 16, kk * 32, DD);
        sacc = WMMA_F32_F16(qa[kk], bT, sacc);
      }
      int n  = lane & 15;
      int kc = wave * 16 + n;
      int kr = krow0 + kc;
      int mb = (lane >= 16) ? 8 : 0;
      float k2v = sk2[kc];
#pragma unroll
      for (int v = 0; v < 8; ++v) {
        int m = mb + v, qr = qrow0 + m;
        float d2 = fmaxf(sq2[m] + k2v - 2.f * sacc[v], 0.f);
        float lg = -TINV * sqrtf(d2) + ((qr == kr) ? lam : 0.f);
        float p  = (kr < LL && qr < LL) ? expf(lg) : 0.f;  // logits <= lam: no max pass
        sP[m * 32 + kc] = (_Float16)p;
      }
    }
    __syncthreads();

    if (tid < 16) {  // softmax denominator accumulation
      float s = 0.f;
      for (int j = 0; j < 32; ++j) s += (float)sP[tid * 32 + j];
      srs[tid] += s;
    }
    { // all waves: H += P @ V (K=32 in one WMMA per 16-col tile)
      v16h a  = load_a_frag_v(sP, 0, 0, 32);
      v16h b0 = load_b_fragK(sKVT, wave * 32, 0, 32);
      acc0 = WMMA_F32_F16(a, b0, acc0);
      v16h b1 = load_b_fragK(sKVT, wave * 32 + 16, 0, 32);
      acc1 = WMMA_F32_F16(a, b1, acc1);
    }
    __syncthreads();
  }

  // ---- normalize + stage H (f32)
  {
    int n = lane & 15, mb = (lane >= 16) ? 8 : 0;
#pragma unroll
    for (int v = 0; v < 8; ++v) {
      int m = mb + v;
      float d   = srs[m];
      float inv = (d > 0.f) ? 1.f / d : 0.f;
      sH[m * DD + wave * 32 + n]      = acc0[v] * inv;
      sH[m * DD + wave * 32 + 16 + n] = acc1[v] * inv;
    }
  }
  __syncthreads();

  // ---- LayerNorm over D, GELU, residual blend, store (half-wave per row)
  {
    int row = (wave << 1) | (lane >> 4);
    int sl  = lane & 15;
    float s = 0.f, s2 = 0.f;
    for (int c = sl; c < DD; c += 16) { float v = sH[row * DD + c]; s += v; s2 += v * v; }
#pragma unroll
    for (int m = 8; m; m >>= 1) { s += __shfl_xor(s, m, 32); s2 += __shfl_xor(s2, m, 32); }
    float mean = s * (1.f / DD);
    float var  = s2 * (1.f / DD) - mean * mean;
    float rstd = rsqrtf(var + LN_EPS);
    int qr = qrow0 + row;
    if (qr < LL) {
      float av = clamp01(alphaP[0]);
      for (int c = sl; c < DD; c += 16) {
        float v  = lng[c] * (sH[row * DD + c] - mean) * rstd + lnb[c];
        float ge = geluf(v);
        size_t idx = ((size_t)bc * LL + qr) * DD + c;
        Xout[idx] = (1.f - av) * Xin[idx] + av * ge;
      }
    }
  }
}

// ==========================================================================
// Kernel 4: fused dim-reduction MLP 256->64->32->16->1 (LN+GELU stages)
// ==========================================================================
__global__ __launch_bounds__(64)
void mlp_kernel(const float* __restrict__ Xin,
                const float* W1, const float* b1, const float* g1, const float* be1,
                const float* W2, const float* b2, const float* g2, const float* be2,
                const float* W3, const float* b3, const float* g3, const float* be3,
                const float* W4, const float* b4, float* __restrict__ Sout) {
  __shared__ float sx[DD];
  __shared__ float h[64];
  __shared__ float rb[64];
  const int row = blockIdx.x, t = threadIdx.x;
  const float* xr = Xin + (size_t)row * DD;
  for (int c = t; c < DD; c += 64) sx[c] = xr[c];
  __syncthreads();

  float mean, rstd;
  // ---- 256 -> 64
  float a1 = b1[t];
  for (int k = 0; k < DD; ++k) a1 = fmaf(sx[k], W1[k * 64 + t], a1);
  rb[t] = a1; __syncthreads();
  for (int s = 32; s; s >>= 1) { if (t < s) rb[t] += rb[t + s]; __syncthreads(); }
  mean = rb[0] * (1.f / 64); __syncthreads();
  { float d = a1 - mean; rb[t] = d * d; } __syncthreads();
  for (int s = 32; s; s >>= 1) { if (t < s) rb[t] += rb[t + s]; __syncthreads(); }
  rstd = rsqrtf(rb[0] * (1.f / 64) + LN_EPS); __syncthreads();
  h[t] = geluf(g1[t] * (a1 - mean) * rstd + be1[t]);
  __syncthreads();
  // ---- 64 -> 32
  float a2 = 0.f;
  if (t < 32) { a2 = b2[t]; for (int k = 0; k < 64; ++k) a2 = fmaf(h[k], W2[k * 32 + t], a2); }
  rb[t] = (t < 32) ? a2 : 0.f; __syncthreads();
  for (int s = 32; s; s >>= 1) { if (t < s) rb[t] += rb[t + s]; __syncthreads(); }
  mean = rb[0] * (1.f / 32); __syncthreads();
  { float d = (t < 32) ? (a2 - mean) : 0.f; rb[t] = d * d; } __syncthreads();
  for (int s = 32; s; s >>= 1) { if (t < s) rb[t] += rb[t + s]; __syncthreads(); }
  rstd = rsqrtf(rb[0] * (1.f / 32) + LN_EPS); __syncthreads();
  if (t < 32) h[t] = geluf(g2[t] * (a2 - mean) * rstd + be2[t]);
  __syncthreads();
  // ---- 32 -> 16
  float a3 = 0.f;
  if (t < 16) { a3 = b3[t]; for (int k = 0; k < 32; ++k) a3 = fmaf(h[k], W3[k * 16 + t], a3); }
  rb[t] = (t < 16) ? a3 : 0.f; __syncthreads();
  for (int s = 32; s; s >>= 1) { if (t < s) rb[t] += rb[t + s]; __syncthreads(); }
  mean = rb[0] * (1.f / 16); __syncthreads();
  { float d = (t < 16) ? (a3 - mean) : 0.f; rb[t] = d * d; } __syncthreads();
  for (int s = 32; s; s >>= 1) { if (t < s) rb[t] += rb[t + s]; __syncthreads(); }
  rstd = rsqrtf(rb[0] * (1.f / 16) + LN_EPS); __syncthreads();
  if (t < 16) h[t] = geluf(g3[t] * (a3 - mean) * rstd + be3[t]);
  __syncthreads();
  // ---- 16 -> 1 (GELU, no LN)
  if (t == 0) {
    float a4 = b4[0];
    for (int k = 0; k < 16; ++k) a4 = fmaf(h[k], W4[k], a4);
    Sout[row] = geluf(a4);
  }
}

// ==========================================================================
// Kernel 5a: img layer linear y = s @ W(500x500) + b
// ==========================================================================
__global__ __launch_bounds__(256)
void img_gemv_kernel(const float* __restrict__ S, const float* __restrict__ W,
                     const float* __restrict__ bias, float* __restrict__ Yimg) {
  __shared__ float sx[LL];
  const int bc = blockIdx.x, t = threadIdx.x;
  for (int l = t; l < LL; l += 256) sx[l] = S[(size_t)bc * LL + l];
  __syncthreads();
  for (int j = t; j < LL; j += 256) {
    float acc = bias[j];
    for (int l = 0; l < LL; ++l) acc = fmaf(sx[l], W[(size_t)l * LL + j], acc);
    Yimg[(size_t)bc * LL + j] = acc;
  }
}

__device__ inline float block_reduce_256(float v, float* rb) {
  int t = threadIdx.x;
  rb[t] = v; __syncthreads();
  for (int s = 128; s; s >>= 1) { if (t < s) rb[t] += rb[t + s]; __syncthreads(); }
  float r = rb[0]; __syncthreads();
  return r;
}

// ==========================================================================
// Kernel 5b: img graph over NC=12 nodes per batch
// ==========================================================================
__global__ __launch_bounds__(256)
void img_graph_kernel(const float* __restrict__ Sin, const float* __restrict__ Yimg,
                      const float* __restrict__ g, const float* __restrict__ be,
                      const float* __restrict__ alphaP, const float* __restrict__ lamdaP,
                      float* __restrict__ Sout) {
  __shared__ float sy[NCC * LL];
  __shared__ float sh[NCC * LL];
  __shared__ float dotm[NCC * NCC];
  __shared__ float adj[NCC * NCC];
  __shared__ float rb[256];
  const int b = blockIdx.x, t = threadIdx.x;

  for (int i = t; i < NCC * LL; i += 256) sy[i] = Yimg[(size_t)b * NCC * LL + i];
  __syncthreads();
  if (t < NCC * NCC) {
    int i = t / NCC, j = t % NCC;
    float d = 0.f;
    for (int l = 0; l < LL; ++l) d = fmaf(sy[i * LL + l], sy[j * LL + l], d);
    dotm[t] = d;
  }
  __syncthreads();
  if (t < NCC) {
    float lam = lamdaP[0];
    float q2 = dotm[t * NCC + t];
    float e[NCC]; float s = 0.f;
    for (int j = 0; j < NCC; ++j) {
      float d2 = fmaxf(q2 + dotm[j * NCC + j] - 2.f * dotm[t * NCC + j], 0.f);
      float lg = -TINV * sqrtf(d2) + ((j == t) ? lam : 0.f);
      e[j] = expf(lg); s += e[j];
    }
    for (int j = 0; j < NCC; ++j) adj[t * NCC + j] = e[j] / s;
  }
  __syncthreads();
  for (int i = t; i < NCC * LL; i += 256) {
    int c = i / LL, l = i % LL;
    float h = 0.f;
    for (int j = 0; j < NCC; ++j) h = fmaf(adj[c * NCC + j], sy[j * LL + l], h);
    sh[i] = h;
  }
  __syncthreads();
  float av = clamp01(alphaP[0]);
  for (int c = 0; c < NCC; ++c) {
    float ps = 0.f, ps2 = 0.f;
    for (int l = t; l < LL; l += 256) { float v = sh[c * LL + l]; ps += v; ps2 += v * v; }
    float sum  = block_reduce_256(ps, rb);
    float sum2 = block_reduce_256(ps2, rb);
    float mean = sum * (1.f / LL);
    float var  = sum2 * (1.f / LL) - mean * mean;
    float rstd = rsqrtf(var + LN_EPS);
    for (int l = t; l < LL; l += 256) {
      float v = g[l] * (sh[c * LL + l] - mean) * rstd + be[l];
      size_t idx = ((size_t)b * NCC + c) * LL + l;
      Sout[idx] = (1.f - av) * Sin[idx] + av * geluf(v);
    }
    __syncthreads();
  }
}

// ==========================================================================
// Kernel 6: cosine global scores
// ==========================================================================
__global__ __launch_bounds__(32)
void gscore_kernel(const float* __restrict__ qg, const float* __restrict__ cg,
                   float* __restrict__ gs) {
  const int p = blockIdx.x;
  const int b = p / (NCC - 1);
  const int lane = threadIdx.x;
  float dq = 0.f, dc = 0.f, dd = 0.f;
  for (int k = lane; k < DD; k += 32) {
    float q = qg[(size_t)b * DD + k];
    float c = cg[(size_t)p * DD + k];
    dq += q * q; dc += c * c; dd += q * c;
  }
#pragma unroll
  for (int off = 16; off; off >>= 1) {
    dq += __shfl_xor(dq, off, 32); dc += __shfl_xor(dc, off, 32); dd += __shfl_xor(dd, off, 32);
  }
  if (lane == 0) {
    float nq = fmaxf(sqrtf(dq), 1e-8f), nc = fmaxf(sqrtf(dc), 1e-8f);
    gs[p] = dd / (nq * nc);
  }
}

// ==========================================================================
// Kernel 7: cls head + softmax + final score. out = [local(352) | final(176)]
// ==========================================================================
__global__ __launch_bounds__(64)
void final_kernel(const float* __restrict__ S, const float* __restrict__ gs,
                  const float* __restrict__ clsW, const float* __restrict__ clsb,
                  const float* __restrict__ ratioP, float* __restrict__ out) {
  __shared__ float lg[NCC * 2];
  const int b = blockIdx.x, t = threadIdx.x;
  if (t < NCC * 2) {
    int c = t >> 1, k = t & 1;
    float acc = clsb[k];
    const float* row = S + ((size_t)b * NCC + c) * LL;
    for (int l = 0; l < LL; ++l) acc = fmaf(row[l], clsW[l * 2 + k], acc);
    lg[t] = acc;
  }
  __syncthreads();
  if (t < NCC - 1) {
    int c = t + 1;
    float l0 = lg[c * 2 + 0], l1 = lg[c * 2 + 1];
    out[((size_t)b * (NCC - 1) + t) * 2 + 0] = l0;
    out[((size_t)b * (NCC - 1) + t) * 2 + 1] = l1;
    float m = fmaxf(l0, l1);
    float e0 = expf(l0 - m), e1 = expf(l1 - m);
    float p1 = e1 / (e0 + e1);
    float r = clamp01(ratioP[0]);
    out[BB * (NCC - 1) * 2 + (size_t)b * (NCC - 1) + t] = gs[b * (NCC - 1) + t] * r + p1 * (1.f - r);
  }
}

// ==========================================================================
// host launcher
// ==========================================================================
extern "C" void kernel_launch(void* const* d_in, const int* in_sizes, int n_in,
                              void* d_out, int out_size, void* d_ws, size_t ws_size,
                              hipStream_t stream) {
  const float* qg       = (const float*)d_in[0];
  const float* cgl      = (const float*)d_in[1];
  const float* xr       = (const float*)d_in[2];
  const float* self_W   = (const float*)d_in[3];
  const float* self_b   = (const float*)d_in[4];
  const float* self_g   = (const float*)d_in[5];
  const float* self_be  = (const float*)d_in[6];
  const float* self_al  = (const float*)d_in[7];
  const float* self_la  = (const float*)d_in[8];
  const float* cross_W  = (const float*)d_in[9];
  const float* cross_b  = (const float*)d_in[10];
  const float* cross_g  = (const float*)d_in[11];
  const float* cross_be = (const float*)d_in[12];
  const float* cross_al = (const float*)d_in[13];
  const float* cross_la = (const float*)d_in[14];
  const float* img_W    = (const float*)d_in[15];
  const float* img_b    = (const float*)d_in[16];
  const float* img_g    = (const float*)d_in[17];
  const float* img_be   = (const float*)d_in[18];
  const float* img_al   = (const float*)d_in[19];
  const float* img_la   = (const float*)d_in[20];
  const float* dr_W1 = (const float*)d_in[21]; const float* dr_b1 = (const float*)d_in[22];
  const float* dr_g1 = (const float*)d_in[23]; const float* dr_be1= (const float*)d_in[24];
  const float* dr_W2 = (const float*)d_in[25]; const float* dr_b2 = (const float*)d_in[26];
  const float* dr_g2 = (const float*)d_in[27]; const float* dr_be2= (const float*)d_in[28];
  const float* dr_W3 = (const float*)d_in[29]; const float* dr_b3 = (const float*)d_in[30];
  const float* dr_g3 = (const float*)d_in[31]; const float* dr_be3= (const float*)d_in[32];
  const float* dr_W4 = (const float*)d_in[33]; const float* dr_b4 = (const float*)d_in[34];
  const float* cls_W = (const float*)d_in[35]; const float* cls_b = (const float*)d_in[36];
  const float* ratio = (const float*)d_in[37];
  float* out = (float*)d_out;

  // workspace carve-up
  float* W = (float*)d_ws;
  size_t o = 0;
  float* gs = W + o; o += 256;
  float* y2 = W + o; o += NROWS;
  float* Yb = W + o; o += (size_t)NROWS * DD;
  float* X0 = W + o; o += (size_t)NROWS * DD;
  float* X1 = W + o; o += (size_t)NROWS * DD;
  float* S0 = W + o; o += NROWS;
  float* S1 = W + o; o += NROWS;
  float* SY = W + o; o += NROWS;
  _Float16* Yh = (_Float16*)(W + o); o += (size_t)NROWS * DD / 2;
  _Float16* Wt = (_Float16*)(W + o); o += (size_t)DD * DD / 2;

  gscore_kernel<<<BB * (NCC - 1), 32, 0, stream>>>(qg, cgl, gs);

  struct Lay { const float *Wp, *bp, *gp, *bep, *al, *la; int cross; };
  Lay lay[4];
  for (int i = 0; i < 2; ++i) {
    lay[2 * i]     = { self_W  + (size_t)i * DD * DD, self_b  + i * DD, self_g  + i * DD,
                       self_be + i * DD, self_al + i, self_la + i, 0 };
    lay[2 * i + 1] = { cross_W + (size_t)i * DD * DD, cross_b + i * DD, cross_g + i * DD,
                       cross_be + i * DD, cross_al + i, cross_la + i, 1 };
  }
  const float* xin = xr;
  float* pong[2] = { X0, X1 };
  for (int li = 0; li < 4; ++li) {
    wcvt_kernel<<<DD * DD / 256, 256, 0, stream>>>(lay[li].Wp, Wt);
    gemm_xw_kernel<<<NROWS / 32, 256, 0, stream>>>(xin, Wt, lay[li].bp, Yb);
    rownorm_kernel<<<NROWS / 8, 256, 0, stream>>>(Yb, y2, Yh);
    attn_kernel<<<BB * NCC * 32, 256, 0, stream>>>(Yh, y2, xin, pong[li & 1],
                                                   lay[li].gp, lay[li].bep,
                                                   lay[li].al, lay[li].la, lay[li].cross);
    xin = pong[li & 1];
  }

  mlp_kernel<<<NROWS, 64, 0, stream>>>(xin,
      dr_W1, dr_b1, dr_g1, dr_be1, dr_W2, dr_b2, dr_g2, dr_be2,
      dr_W3, dr_b3, dr_g3, dr_be3, dr_W4, dr_b4, S0);

  img_gemv_kernel<<<BB * NCC, 256, 0, stream>>>(S0, img_W, img_b, SY);
  img_graph_kernel<<<BB, 256, 0, stream>>>(S0, SY, img_g, img_be, img_al, img_la, S1);
  img_gemv_kernel<<<BB * NCC, 256, 0, stream>>>(S1, img_W + (size_t)LL * LL, img_b + LL, SY);
  img_graph_kernel<<<BB, 256, 0, stream>>>(S1, SY, img_g + LL, img_be + LL,
                                           img_al + 1, img_la + 1, S0);

  final_kernel<<<BB, 64, 0, stream>>>(S0, gs, cls_W, cls_b, ratio, out);
}
